// SimpleRNN_13005160972577
// MI455X (gfx1250) — matrix operand order repaired and checked by reading, not compiled, not run on hardware
//
#include <hip/hip_runtime.h>

#define RNN_B 32
#define RNN_S 2048
#define RNN_I 256
#define RNN_H 512
#define RNN_O 256
#define NWG   8

typedef __attribute__((ext_vector_type(16))) __bf16         bf16x16;
typedef __attribute__((ext_vector_type(8)))  float          f32x8;
typedef __attribute__((ext_vector_type(8)))  unsigned short u16x8;

union frag_cast { bf16x16 v; u16x8 h[2]; };

__device__ __forceinline__ unsigned short f2bf(float f) {
  union { float f; unsigned u; } x; x.f = f;
  unsigned r = x.u + 0x7FFFu + ((x.u >> 16) & 1u);   // round-to-nearest-even
  return (unsigned short)(r >> 16);
}

// A fragment: 16(M) x 32(K) bf16 tile from a row-major matrix (ld elems/row).
// ISA layout: lanes l and l+16 both hold row (l&15); lanes 0-15 carry
// K={0..7,16..23}, lanes 16-31 carry K={8..15,24..31}.
__device__ __forceinline__ bf16x16 load_frag_a(const unsigned short* p, int ld, int lane) {
  const unsigned short* row = p + (size_t)(lane & 15) * ld;
  const int hi = (lane >> 4) & 1;
  frag_cast u;
  u.h[0] = *(const u16x8*)(row + hi * 8);        // K 0..7  or 8..15
  u.h[1] = *(const u16x8*)(row + 16 + hi * 8);   // K 16..23 or 24..31
  return u.v;
}

// B fragment: 32(K) x 16(N) tile; we store B transposed (N x K row-major) so
// each lane reads its own output column. Lanes 0-15: col=lane, K={0..7,16..23};
// lanes 16-31: col=lane-16, K={8..15,24..31}.
__device__ __forceinline__ bf16x16 load_frag_b(const unsigned short* p, int ld, int lane) {
  const unsigned short* row = p + (size_t)(lane & 15) * ld;
  const int hi = (lane >> 4) & 1;
  frag_cast u;
  u.h[0] = *(const u16x8*)(row + hi * 8);
  u.h[1] = *(const u16x8*)(row + 16 + hi * 8);
  return u.v;
}

__device__ __forceinline__ f32x8 wmma_bf16(bf16x16 a, bf16x16 b, f32x8 c) {
  return __builtin_amdgcn_wmma_f32_16x16x32_bf16(false, a, false, b, (short)0, c, false, false);
}

// ---------------------------------------------------------------- utilities
__global__ void f32_to_bf16_kernel(const float* __restrict__ src,
                                   unsigned short* __restrict__ dst, int n) {
  int i = blockIdx.x * blockDim.x + threadIdx.x;
  if (i < n) dst[i] = f2bf(src[i]);
}

__global__ void rnn_init_kernel(unsigned short* hbuf, int* bar) {
  int i = blockIdx.x * blockDim.x + threadIdx.x;
  if (i < 2 * RNN_B * RNN_H) hbuf[i] = 0;       // h0 = 0 (both buffers)
  if (i < 2) bar[i] = 0;                        // barrier count + generation
}

// ------------------------------------------------- input projection (GEMM)
// u_proj[B*S, H] = u_bf16[B*S, I] @ W_ih_bf16[H, I]^T  (fp32 accum)
__global__ void __launch_bounds__(256)
rnn_input_proj(const unsigned short* __restrict__ u_bf,
               const unsigned short* __restrict__ Wih_bf,
               float* __restrict__ uproj) {
  const int wave = threadIdx.x >> 5, lane = threadIdx.x & 31;
  const int tile  = blockIdx.x * 8 + wave;      // 131072 16x16 tiles total
  const int mtile = tile >> 5;                  // 4096 row tiles (B*S/16)
  const int ntile = tile & 31;                  // 32 col tiles (H/16)
  const unsigned short* A  = u_bf  + (size_t)mtile * 16 * RNN_I;
  const unsigned short* Bp = Wih_bf + (size_t)ntile * 16 * RNN_I;   // N x K

  f32x8 c = {0.f, 0.f, 0.f, 0.f, 0.f, 0.f, 0.f, 0.f};
#pragma unroll
  for (int k0 = 0; k0 < RNN_I; k0 += 32)
    c = wmma_bf16(load_frag_a(A + k0, RNN_I, lane),
                  load_frag_b(Bp + k0, RNN_I, lane), c);

  const int r15 = lane & 15, hi = (lane >> 4) & 1;
#pragma unroll
  for (int r = 0; r < 8; ++r) {
    int row = mtile * 16 + r + 8 * hi;          // C layout: lane>=16 -> M+8
    uproj[(size_t)row * RNN_H + ntile * 16 + r15] = c[r];
  }
}

// ------------------------------------------------------ grid-wide barrier
__device__ __forceinline__ void grid_barrier(int* cnt, int* gen, int phase) {
  __threadfence();                               // release h writes
  __syncthreads();
  if (threadIdx.x == 0) {
    if (atomicAdd(cnt, 1) == NWG - 1) {
      atomicExch(cnt, 0);
      __hip_atomic_fetch_add(gen, 1, __ATOMIC_RELEASE, __HIP_MEMORY_SCOPE_AGENT);
    } else {
      while (__hip_atomic_load(gen, __ATOMIC_ACQUIRE, __HIP_MEMORY_SCOPE_AGENT) < phase)
        __builtin_amdgcn_s_sleep(2);
    }
  }
  __syncthreads();
  __threadfence();                               // acquire other WGs' h writes
}

// ------------------------------------------------------------- recurrence
// Persistent grid of NWG=8 workgroups, 8 waves each. WG w owns h columns
// [w*64, w*64+64) and y columns [w*32, w*32+32). W slices cached in LDS.
__global__ void __launch_bounds__(256)
rnn_recurrence(const unsigned short* __restrict__ Whh_bf,   // [H x H] (N x K)
               const unsigned short* __restrict__ Who_bf,   // [O x H] (N x K)
               const float* __restrict__ uproj,             // [B*S x H]
               unsigned short* __restrict__ hbuf,            // [2][B x H] bf16
               float* __restrict__ out,                      // [B,S,O]
               int* bar_cnt, int* bar_gen) {
  __shared__ __attribute__((aligned(16))) unsigned short sWhh[64 * RNN_H]; // 64 KB
  __shared__ __attribute__((aligned(16))) unsigned short sWho[32 * RNN_H]; // 32 KB

  const int wg = blockIdx.x;
  const int tid = threadIdx.x;
  const int wave = tid >> 5, lane = tid & 31;
  const int r15 = lane & 15, hi = (lane >> 4) & 1;

  // Stage this WG's weight slices into LDS once.
  for (int i = tid; i < 64 * RNN_H; i += 256) sWhh[i] = Whh_bf[(size_t)(wg * 64) * RNN_H + i];
  for (int i = tid; i < 32 * RNN_H; i += 256) sWho[i] = Who_bf[(size_t)(wg * 32) * RNN_H + i];
  __syncthreads();

  // h-phase: 8 tiles per WG -> 1 per wave. 2 row tiles x 4 col tiles.
  const int mt = wave & 1;                       // batch-row tile (0..1)
  const int ct = wave >> 1;                      // local col tile (0..3)
  const int hcol = wg * 64 + ct * 16;
  const unsigned short* sB_h = sWhh + (size_t)(ct * 16) * RNN_H;

  // y-phase: 4 tiles per WG -> waves 0..3. 2 row tiles x 2 col tiles.
  const int mt2 = wave & 1;
  const int ct2 = (wave >> 1) & 1;
  const int ycol = wg * 32 + ct2 * 16;
  const unsigned short* sB_y = sWho + (size_t)(ct2 * 16) * RNN_H;

  int phase = 0;
  for (int t = 0; t < RNN_S; ++t) {
    const unsigned short* hprev = hbuf + (size_t)(t & 1) * RNN_B * RNN_H;
    unsigned short*       hcur  = hbuf + (size_t)((t + 1) & 1) * RNN_B * RNN_H;

    if (t + 1 < RNN_S) {   // prefetch next step's uproj slice (global_prefetch_b8)
      const float* pf = uproj + ((size_t)(tid >> 4) * RNN_S + (t + 1)) * RNN_H
                              + (size_t)(tid & 15) * 32;
      __builtin_prefetch(pf, 0, 1);
    }

    // h_t[:, mycols] = tanh(h_{t-1} @ Whh^T + uproj[:, t, mycols])
    f32x8 c = {0.f, 0.f, 0.f, 0.f, 0.f, 0.f, 0.f, 0.f};
    const unsigned short* Arow = hprev + (size_t)(mt * 16) * RNN_H;
#pragma unroll 4
    for (int k0 = 0; k0 < RNN_H; k0 += 32)
      c = wmma_bf16(load_frag_a(Arow + k0, RNN_H, lane),
                    load_frag_b(sB_h + k0, RNN_H, lane), c);

    const int ncol = hcol + r15;
#pragma unroll
    for (int r = 0; r < 8; ++r) {
      int brow = mt * 16 + r + 8 * hi;
      float v = tanhf(c[r] + uproj[((size_t)brow * RNN_S + t) * RNN_H + ncol]);
      hcur[(size_t)brow * RNN_H + ncol] = f2bf(v);
    }

    grid_barrier(bar_cnt, bar_gen, ++phase);     // full h_t now visible

    if (wave < 4) {                              // uniform per wave: EXEC all-1s inside
      f32x8 cy = {0.f, 0.f, 0.f, 0.f, 0.f, 0.f, 0.f, 0.f};
      const unsigned short* Ay = hcur + (size_t)(mt2 * 16) * RNN_H;
#pragma unroll 4
      for (int k0 = 0; k0 < RNN_H; k0 += 32)
        cy = wmma_bf16(load_frag_a(Ay + k0, RNN_H, lane),
                       load_frag_b(sB_y + k0, RNN_H, lane), cy);
      const int oc = ycol + r15;
#pragma unroll
      for (int r = 0; r < 8; ++r) {
        int brow = mt2 * 16 + r + 8 * hi;
        out[((size_t)brow * RNN_S + t) * RNN_O + oc] = cy[r];
      }
    }
  }
}

// ----------------------------------------------------------------- launch
extern "C" void kernel_launch(void* const* d_in, const int* in_sizes, int n_in,
                              void* d_out, int out_size, void* d_ws, size_t ws_size,
                              hipStream_t stream) {
  const float* u   = (const float*)d_in[0];   // [32,2048,256]
  const float* Whh = (const float*)d_in[1];   // [512,512]
  const float* Wih = (const float*)d_in[2];   // [512,256]
  const float* Who = (const float*)d_in[3];   // [256,512]
  float* out = (float*)d_out;                 // [32,2048,256]

  char* ws = (char*)d_ws;
  const size_t N_U   = (size_t)RNN_B * RNN_S * RNN_I;      // 16,777,216
  const size_t N_WHH = (size_t)RNN_H * RNN_H;              //    262,144
  const size_t N_WIH = (size_t)RNN_H * RNN_I;              //    131,072
  const size_t N_WHO = (size_t)RNN_O * RNN_H;              //    131,072

  unsigned short* u_bf   = (unsigned short*)(ws);                       // 32 MB
  unsigned short* Whh_bf = (unsigned short*)(ws + 2 * N_U);
  unsigned short* Wih_bf = (unsigned short*)(ws + 2 * (N_U + N_WHH));
  unsigned short* Who_bf = (unsigned short*)(ws + 2 * (N_U + N_WHH + N_WIH));
  char* after_w = ws + 2 * (N_U + N_WHH + N_WIH + N_WHO);
  float* uproj  = (float*)after_w;                                      // 128 MB
  unsigned short* hbuf = (unsigned short*)(after_w + 4 * (size_t)RNN_B * RNN_S * RNN_H);
  int* bar = (int*)(hbuf + 2 * RNN_B * RNN_H);

  f32_to_bf16_kernel<<<(int)((N_U   + 255) / 256), 256, 0, stream>>>(u,   u_bf,   (int)N_U);
  f32_to_bf16_kernel<<<(int)((N_WHH + 255) / 256), 256, 0, stream>>>(Whh, Whh_bf, (int)N_WHH);
  f32_to_bf16_kernel<<<(int)((N_WIH + 255) / 256), 256, 0, stream>>>(Wih, Wih_bf, (int)N_WIH);
  f32_to_bf16_kernel<<<(int)((N_WHO + 255) / 256), 256, 0, stream>>>(Who, Who_bf, (int)N_WHO);
  rnn_init_kernel<<<128, 256, 0, stream>>>(hbuf, bar);

  // 4096 M-tiles x 32 N-tiles = 131072 tiles, 8 waves/WG -> 16384 WGs
  rnn_input_proj<<<16384, 256, 0, stream>>>(u_bf, Wih_bf, uproj);

  rnn_recurrence<<<NWG, 256, 0, stream>>>(Whh_bf, Who_bf, uproj, hbuf, out,
                                          bar, bar + 1);
}
